// StatefulPrefillFullAttn_39376260170065
// MI455X (gfx1250) — matrix operand order repaired and checked by reading, not compiled, not run on hardware
//
#include <hip/hip_runtime.h>
#include <hip/hip_bf16.h>
#include <math.h>

#define HID    2048
#define NH     16
#define NKV    2
#define HDIM   256
#define SEQ    2048
#define MAXSEQ 4096
#define RDIM   64
#define GROUPS (NH / NKV)

typedef __attribute__((ext_vector_type(16))) __bf16         v16bf;
typedef __attribute__((ext_vector_type(8)))  float          v8f;
typedef __attribute__((ext_vector_type(8)))  unsigned short u16x8;
typedef __attribute__((ext_vector_type(16))) unsigned short u16x16;

union BF16Frag { u16x16 u; v16bf b; };

__device__ __forceinline__ unsigned short f2bf(float f) {
    unsigned u = __float_as_uint(f);
    u += 0x7FFFu + ((u >> 16) & 1u);      // round-to-nearest-even
    return (unsigned short)(u >> 16);
}

__device__ __forceinline__ v8f wmma_bf16(const BF16Frag& a, const BF16Frag& b, v8f c) {
    return __builtin_amdgcn_wmma_f32_16x16x32_bf16(false, a.b, false, b.b,
                                                   (short)0, c, false, false);
}

// Async global->LDS 16-byte copy (CDNA5 GLOBAL_LOAD_ASYNC_TO_LDS_B128, ASYNCcnt).
__device__ __forceinline__ void async_copy_b128(unsigned lds_off, const void* gaddr) {
    unsigned long long ga = (unsigned long long)gaddr;
    asm volatile("global_load_async_to_lds_b128 %0, %1, off"
                 :: "v"(lds_off), "v"(ga) : "memory");
}
template <int N>
__device__ __forceinline__ void wait_async() {
    asm volatile("s_wait_asynccnt %0" :: "i"(N) : "memory");
}

// -------------------------------------------------------------------------
// f32 -> bf16 pack, 8 elems/thread (n must be a multiple of 2048)
// -------------------------------------------------------------------------
__global__ __launch_bounds__(256)
void f32_to_bf16_kernel(const float* __restrict__ in, unsigned short* __restrict__ out,
                        size_t n) {
    size_t i = ((size_t)blockIdx.x * 256 + threadIdx.x) * 8;
    if (i >= n) return;
    const float4* p = (const float4*)(in + i);
    float4 a = p[0], b = p[1];
    u16x8 r;
    r[0] = f2bf(a.x); r[1] = f2bf(a.y); r[2] = f2bf(a.z); r[3] = f2bf(a.w);
    r[4] = f2bf(b.x); r[5] = f2bf(b.y); r[6] = f2bf(b.z); r[7] = f2bf(b.w);
    *(u16x8*)(out + i) = r;
}

// -------------------------------------------------------------------------
// C[M,N] = A[M,K] * W[N,K]^T  with bf16 inputs, f32 output.
// Block: 256 threads = 8 waves. Tile: 128(M) x 64(N), K step 32.
// Double-buffered async global->LDS staging overlapped with WMMA.
// -------------------------------------------------------------------------
__global__ __launch_bounds__(256)
void gemm_bf16_wmma(const unsigned short* __restrict__ A,
                    const unsigned short* __restrict__ W,
                    float* __restrict__ C, int M, int N, int K) {
    __shared__ __align__(32) unsigned short lds_a[2][128 * 32];   // 2 x 8 KB
    __shared__ __align__(32) unsigned short lds_b[2][64 * 32];    // 2 x 4 KB

    const int tid  = threadIdx.x;
    const int wave = tid >> 5;
    const int lane = tid & 31;
    const int half = lane >> 4;
    const int lr   = lane & 15;
    const int m0   = blockIdx.y * 128;
    const int n0   = blockIdx.x * 64;

    auto issue_tile = [&](int buf, int k0) {
        unsigned abase = (unsigned)(unsigned long long)&lds_a[buf][0];
        #pragma unroll
        for (int t = 0; t < 2; ++t) {
            int j = tid + t * 256;           // 0..511 covers 128x32 in 16B units
            int r = j >> 2, c8 = (j & 3) * 8;
            async_copy_b128(abase + (unsigned)j * 16,
                            A + (size_t)(m0 + r) * K + k0 + c8);
        }
        unsigned bbase = (unsigned)(unsigned long long)&lds_b[buf][0];
        int r = tid >> 2, c8 = (tid & 3) * 8;
        async_copy_b128(bbase + (unsigned)tid * 16,
                        W + (size_t)(n0 + r) * K + k0 + c8);
    };

    v8f acc[4] = {};

    issue_tile(0, 0);                         // prologue
    for (int k0 = 0; k0 < K; k0 += 32) {
        const int cur = (k0 >> 5) & 1;
        const bool more = (k0 + 32 < K);
        if (more) {
            issue_tile(1 - cur, k0 + 32);     // stream next tile during compute
            __builtin_prefetch(&W[(size_t)(n0 + (tid >> 2)) * K + k0 + 64], 0, 0);
            wait_async<3>();                  // current tile landed; next in flight
        } else {
            wait_async<0>();
        }
        __syncthreads();

        // A fragment: row = wave*16 + lr ; lane<16 -> K 0..15, lane>=16 -> K 16..31
        const int arow = wave * 16 + lr;
        const int ak   = half * 16;
        u16x8 alo = *(const u16x8*)&lds_a[cur][arow * 32 + ak];
        u16x8 ahi = *(const u16x8*)&lds_a[cur][arow * 32 + ak + 8];
        BF16Frag af;
        af.u = __builtin_shufflevector(alo, ahi, 0,1,2,3,4,5,6,7,8,9,10,11,12,13,14,15);

        #pragma unroll
        for (int t = 0; t < 4; ++t) {
            // B fragment: column n = t*16 + lr ; lane<16 -> K 0..15, lane>=16 -> K 16..31
            BF16Frag bfr;
            bfr.u = *(const u16x16*)&lds_b[cur][(t * 16 + lr) * 32 + half * 16];
            acc[t] = wmma_bf16(af, bfr, acc[t]);
        }
        __syncthreads();
    }

    // C layout: VGPR i -> row i + 8*half, col = lane&15 within tile
    #pragma unroll
    for (int t = 0; t < 4; ++t) {
        int gn = n0 + t * 16 + lr;
        #pragma unroll
        for (int i = 0; i < 8; ++i) {
            int gm = m0 + wave * 16 + half * 8 + i;
            C[(size_t)gm * N + gn] = acc[t][i];
        }
    }
}

// -------------------------------------------------------------------------
// RMSNorm + RoPE. Grid (SEQ, NH+NKV), block HDIM threads.
// Also emits bf16 K cache ([kv][s][d]) and transposed bf16 V cache ([kv][d][s])
// so attention can stage tiles with pure async b128 copies.
// -------------------------------------------------------------------------
__global__ __launch_bounds__(HDIM)
void postproc_norm_rope(const float* __restrict__ cosb, const float* __restrict__ sinb,
                        float* __restrict__ qg, const float* __restrict__ kin,
                        const float* __restrict__ vin,
                        const float* __restrict__ qnw, const float* __restrict__ knw,
                        float* __restrict__ kcache, float* __restrict__ vcache,
                        unsigned short* __restrict__ k_bf,
                        unsigned short* __restrict__ vT_bf) {
    __shared__ float red[HDIM];
    __shared__ float shy[HDIM];
    const int s = blockIdx.x;
    const int h = blockIdx.y;
    const int d = threadIdx.x;

    float x = (h < NH) ? qg[(size_t)s * 8192 + h * 512 + d]
                       : kin[(size_t)s * 512 + (h - NH) * HDIM + d];
    red[d] = x * x;
    __syncthreads();
    for (int off = HDIM / 2; off > 0; off >>= 1) {
        if (d < off) red[d] += red[d + off];
        __syncthreads();
    }
    float rms = rsqrtf(red[0] / (float)HDIM + 1e-6f);
    float w   = (h < NH) ? qnw[d] : knw[d];
    float y   = x * rms * (1.0f + w);
    shy[d] = y;
    __syncthreads();

    float out = y;
    if (d < RDIM) {
        float c  = cosb[(size_t)s * RDIM + d];
        float sn = sinb[(size_t)s * RDIM + d];
        float rh = (d < RDIM / 2) ? -shy[d + RDIM / 2] : shy[d - RDIM / 2];
        out = y * c + rh * sn;
    }
    if (h < NH) {
        qg[(size_t)s * 8192 + h * 512 + d] = out;
    } else {
        int hk = h - NH;
        float vv = vin[(size_t)s * 512 + hk * HDIM + d];
        kcache[((size_t)hk * MAXSEQ + s) * HDIM + d] = out;
        vcache[((size_t)hk * MAXSEQ + s) * HDIM + d] = vv;
        k_bf[((size_t)hk * SEQ + s) * HDIM + d]      = f2bf(out);
        vT_bf[((size_t)hk * HDIM + d) * SEQ + s]     = f2bf(vv);
    }
}

__global__ void pad_zero(float* __restrict__ kcache, float* __restrict__ vcache) {
    size_t i = (size_t)blockIdx.x * 256 + threadIdx.x;
    const size_t per = (size_t)(MAXSEQ - SEQ) * HDIM;
    int hk = (int)(i / per);
    size_t rem = i % per;
    size_t off = ((size_t)hk * MAXSEQ + SEQ) * HDIM + rem;
    kcache[off] = 0.0f;
    vcache[off] = 0.0f;
}

// -------------------------------------------------------------------------
// Flash attention with WMMA. Grid (SEQ/128, NH), block 256 (8 waves).
// K and V^T tiles double-buffered via async global->LDS copies (bf16 caches).
// Output written directly as bf16 for the o-projection GEMM.
// -------------------------------------------------------------------------
__global__ __launch_bounds__(256)
void flash_attn_wmma(const float* __restrict__ qg,
                     const unsigned short* __restrict__ k_bf,
                     const unsigned short* __restrict__ vT_bf,
                     unsigned short* __restrict__ attn_bf) {
    __shared__ __align__(32) unsigned short lds_k[2][32 * HDIM];    // 2 x 16 KB [key][d]
    __shared__ __align__(32) unsigned short lds_vt[2][HDIM * 32];   // 2 x 16 KB [d][key]
    __shared__ __align__(32) unsigned short lds_p[8][16 * 32];      // per-wave P tile

    const int tid  = threadIdx.x;
    const int wave = tid >> 5;
    const int lane = tid & 31;
    const int half = lane >> 4;
    const int lr   = lane & 15;
    const int h    = blockIdx.y;
    const int kv   = h / GROUPS;
    const int qblk = blockIdx.x * 128;
    const int qbase = qblk + wave * 16;
    const float scale = rsqrtf((float)HDIM);

    const unsigned short* kc  = k_bf + (size_t)kv * SEQ * HDIM;
    const unsigned short* vtc = vT_bf + (size_t)kv * HDIM * SEQ;

    auto issue_kv = [&](int buf, int kbase) {
        unsigned kb = (unsigned)(unsigned long long)&lds_k[buf][0];
        unsigned vb = (unsigned)(unsigned long long)&lds_vt[buf][0];
        #pragma unroll
        for (int t = 0; t < 4; ++t) {
            int j = tid + t * 256;                 // 0..1023 (16B units)
            int key = j >> 5, o16 = (j & 31) * 8;  // K tile: 32 x 256
            async_copy_b128(kb + (unsigned)j * 16,
                            kc + (size_t)(kbase + key) * HDIM + o16);
            int d = j >> 2, c8 = (j & 3) * 8;      // V^T tile: 256 x 32
            async_copy_b128(vb + (unsigned)j * 16,
                            vtc + (size_t)d * SEQ + kbase + c8);
        }
    };

    // Q fragments over 8 chunks of K=32 along head dim (pre-scaled)
    BF16Frag qf[8];
    {
        const float* qrow = qg + (size_t)(qbase + lr) * 8192 + h * 512;
        #pragma unroll
        for (int c = 0; c < 8; ++c) {
            int k1 = c * 32 + half * 16;
            #pragma unroll
            for (int j = 0; j < 8; ++j) {
                qf[c].u[j]     = f2bf(qrow[k1 + j] * scale);
                qf[c].u[j + 8] = f2bf(qrow[k1 + 8 + j] * scale);
            }
        }
    }

    v8f o[16] = {};
    float mstat[8], lstat[8];
    #pragma unroll
    for (int i = 0; i < 8; ++i) { mstat[i] = -1e30f; lstat[i] = 0.0f; }

    const int kend = qblk + 128;
    issue_kv(0, 0);                                // prologue
    for (int kbase = 0; kbase < kend; kbase += 32) {
        const int cur = (kbase >> 5) & 1;
        const bool more = (kbase + 32 < kend);
        if (more) {
            issue_kv(1 - cur, kbase + 32);         // stream next chunk during compute
            wait_async<8>();
        } else {
            wait_async<0>();
        }
        __syncthreads();

        if (kbase <= qbase + 15) {
            v8f s0 = {}, s1 = {};
            #pragma unroll
            for (int c = 0; c < 8; ++c) {
                BF16Frag b0, b1;                   // B = K^T: col n = key, K dim = d
                b0.u = *(const u16x16*)&lds_k[cur][lr * HDIM + c * 32 + half * 16];
                b1.u = *(const u16x16*)&lds_k[cur][(lr + 16) * HDIM + c * 32 + half * 16];
                s0 = wmma_bf16(qf[c], b0, s0);
                s1 = wmma_bf16(qf[c], b1, s1);
            }

            #pragma unroll
            for (int i = 0; i < 8; ++i) {
                int gq = qbase + half * 8 + i;
                if (kbase + lr > gq)      s0[i] += -10000.0f;
                if (kbase + 16 + lr > gq) s1[i] += -10000.0f;

                float mx = fmaxf(s0[i], s1[i]);
                for (int m = 8; m >= 1; m >>= 1) mx = fmaxf(mx, __shfl_xor(mx, m, 16));
                float nm = fmaxf(mstat[i], mx);
                float p0 = __expf(s0[i] - nm);
                float p1 = __expf(s1[i] - nm);
                s0[i] = p0; s1[i] = p1;
                float ps = p0 + p1;
                for (int m = 8; m >= 1; m >>= 1) ps += __shfl_xor(ps, m, 16);
                float f = __expf(mstat[i] - nm);
                lstat[i] = lstat[i] * f + ps;
                mstat[i] = nm;
                #pragma unroll
                for (int t = 0; t < 16; ++t) o[t][i] *= f;
            }

            #pragma unroll
            for (int i = 0; i < 8; ++i) {
                int row = half * 8 + i;
                lds_p[wave][row * 32 + lr]      = f2bf(s0[i]);
                lds_p[wave][row * 32 + lr + 16] = f2bf(s1[i]);
            }
            BF16Frag pf;
            {
                int ak = half * 16;
                u16x8 lo = *(const u16x8*)&lds_p[wave][lr * 32 + ak];
                u16x8 hi = *(const u16x8*)&lds_p[wave][lr * 32 + ak + 8];
                pf.u = __builtin_shufflevector(lo, hi, 0,1,2,3,4,5,6,7,8,9,10,11,12,13,14,15);
            }
            #pragma unroll
            for (int t = 0; t < 16; ++t) {         // B = V: col n = d, K = key
                BF16Frag vf;
                vf.u = *(const u16x16*)&lds_vt[cur][(t * 16 + lr) * 32 + half * 16];
                o[t] = wmma_bf16(pf, vf, o[t]);
            }
        }
        __syncthreads();
    }

    // epilogue: normalize + fused sigmoid gate, emit bf16 for o-proj GEMM
    #pragma unroll
    for (int t = 0; t < 16; ++t) {
        int d = t * 16 + lr;
        #pragma unroll
        for (int i = 0; i < 8; ++i) {
            int row = qbase + half * 8 + i;
            float g   = qg[(size_t)row * 8192 + h * 512 + HDIM + d];
            float sig = 1.0f / (1.0f + __expf(-g));
            attn_bf[(size_t)row * 4096 + h * HDIM + d] = f2bf((o[t][i] / lstat[i]) * sig);
        }
    }
}

extern "C" void kernel_launch(void* const* d_in, const int* in_sizes, int n_in,
                              void* d_out, int out_size, void* d_ws, size_t ws_size,
                              hipStream_t stream) {
    const float* hidden = (const float*)d_in[0];
    const float* cosb   = (const float*)d_in[1];
    const float* sinb   = (const float*)d_in[2];
    const float* qw     = (const float*)d_in[3];
    const float* kw     = (const float*)d_in[4];
    const float* vw     = (const float*)d_in[5];
    const float* ow     = (const float*)d_in[6];
    const float* qnw    = (const float*)d_in[7];
    const float* knw    = (const float*)d_in[8];

    float* out        = (float*)d_out;
    float* hidden_out = out;                                    // SEQ*HID
    float* kcache     = out + (size_t)SEQ * HID;                // NKV*MAXSEQ*HDIM
    float* vcache     = kcache + (size_t)NKV * MAXSEQ * HDIM;

    // ---- workspace layout ----
    float* ws_f    = (float*)d_ws;
    float* qg_ws   = ws_f;                                      // SEQ*8192 f32
    float* k_ws    = qg_ws + (size_t)SEQ * 2 * NH * HDIM;       // SEQ*512 f32
    float* v_ws    = k_ws + (size_t)SEQ * NKV * HDIM;           // SEQ*512 f32
    unsigned short* ws_h    = (unsigned short*)(v_ws + (size_t)SEQ * NKV * HDIM);
    unsigned short* hid_bf  = ws_h;                             // SEQ*HID
    unsigned short* qw_bf   = hid_bf + (size_t)SEQ * HID;       // 8192*HID
    unsigned short* kw_bf   = qw_bf + (size_t)2 * NH * HDIM * HID;
    unsigned short* vw_bf   = kw_bf + (size_t)NKV * HDIM * HID;
    unsigned short* ow_bf   = vw_bf + (size_t)NKV * HDIM * HID; // HID*(NH*HDIM)
    unsigned short* attn_bf = ow_bf + (size_t)HID * NH * HDIM;  // SEQ*(NH*HDIM)
    unsigned short* k_bf    = attn_bf + (size_t)SEQ * NH * HDIM; // NKV*SEQ*HDIM
    unsigned short* vT_bf   = k_bf + (size_t)NKV * SEQ * HDIM;   // NKV*HDIM*SEQ

    auto cvt = [&](const float* src, unsigned short* dst, size_t n) {
        f32_to_bf16_kernel<<<dim3((unsigned)(n / 2048)), 256, 0, stream>>>(src, dst, n);
    };
    // 0) one-time f32 -> bf16 packs (activations + weights)
    cvt(hidden, hid_bf, (size_t)SEQ * HID);
    cvt(qw, qw_bf, (size_t)2 * NH * HDIM * HID);
    cvt(kw, kw_bf, (size_t)NKV * HDIM * HID);
    cvt(vw, vw_bf, (size_t)NKV * HDIM * HID);
    cvt(ow, ow_bf, (size_t)HID * NH * HDIM);

    // 1) projections: q+gate, k, v (double-buffered async bf16 WMMA GEMMs)
    gemm_bf16_wmma<<<dim3((2 * NH * HDIM) / 64, SEQ / 128), 256, 0, stream>>>(
        hid_bf, qw_bf, qg_ws, SEQ, 2 * NH * HDIM, HID);
    gemm_bf16_wmma<<<dim3((NKV * HDIM) / 64, SEQ / 128), 256, 0, stream>>>(
        hid_bf, kw_bf, k_ws, SEQ, NKV * HDIM, HID);
    gemm_bf16_wmma<<<dim3((NKV * HDIM) / 64, SEQ / 128), 256, 0, stream>>>(
        hid_bf, vw_bf, v_ws, SEQ, NKV * HDIM, HID);

    // 2) rmsnorm + rope (q in-place; f32 caches to d_out, bf16 K / V^T for attention)
    postproc_norm_rope<<<dim3(SEQ, NH + NKV), HDIM, 0, stream>>>(
        cosb, sinb, qg_ws, k_ws, v_ws, qnw, knw, kcache, vcache, k_bf, vT_bf);

    // 3) zero cache padding rows [SEQ, MAXSEQ)
    pad_zero<<<dim3((NKV * (MAXSEQ - SEQ) * HDIM) / 256), 256, 0, stream>>>(kcache, vcache);

    // 4) flash attention with fused sigmoid gate (async-staged bf16 K/V^T)
    flash_attn_wmma<<<dim3(SEQ / 128, NH), 256, 0, stream>>>(qg_ws, k_bf, vT_bf, attn_bf);

    // 5) output projection
    gemm_bf16_wmma<<<dim3(HID / 64, SEQ / 128), 256, 0, stream>>>(
        attn_bf, ow_bf, hidden_out, SEQ, HID, NH * HDIM);
}
